// GRU_86354612453408
// MI455X (gfx1250) — matrix operand order repaired and checked by reading, not compiled, not run on hardware
//
#include <hip/hip_runtime.h>

// ---------------------------------------------------------------------------
// Bidirectional 2-layer GRU for MI455X (gfx1250), bf16 WMMA with fp32 state.
//   Phase A: x-projections hoisted to big GEMMs (M=16384, N=1536) via
//            v_wmma_f32_16x16x32_bf16 (pipelined global_load_b128 clauses).
//   Phase B: ONE persistent kernel per layer; block = direction; the whole
//            T=512 scan runs inside the block with recurrent state (h fp32,
//            z fp32, h bf16, r*h bf16 = 192KB) held in LDS (320KB/WGP).
// Recurrent weights (1.5MB bf16/cell) stay L2-resident (192MB L2).
// ---------------------------------------------------------------------------

#define T_SEQ  512
#define NBATCH 32
#define HID    512

typedef __attribute__((ext_vector_type(16))) __bf16 v16bf;
typedef __attribute__((ext_vector_type(8)))  __bf16 v8bf;
typedef __attribute__((ext_vector_type(8)))  float  v8f;

__device__ __forceinline__ v8f wmma_bf16(v16bf a, v16bf b, v8f c) {
  return __builtin_amdgcn_wmma_f32_16x16x32_bf16(false, a, false, b, (short)0, c,
                                                 false, false);
}

// A-matrix fragment: 16x32 bf16 tile, row-major (ld elements per row).
// Layout: lanes0-15 K={kb..kb+15}, lanes16-31 K={kb+16..kb+31} (matches load_b).
__device__ __forceinline__ v16bf load_a(const __bf16* __restrict__ base, int ld,
                                        int row, int kb, int r, int h) {
  const __bf16* p = base + (size_t)(row + r) * (size_t)ld + kb + h * 16;
  v8bf lo = *(const v8bf*)p;
  v8bf hi = *(const v8bf*)(p + 8);
  return __builtin_shufflevector(lo, hi, 0, 1, 2, 3, 4, 5, 6, 7, 8, 9, 10, 11,
                                 12, 13, 14, 15);
}

// B-matrix fragment from transposed (N x K) storage: lane n = L%16,
// half L/16 selects K block of 16 -> one contiguous 32B span per lane.
__device__ __forceinline__ v16bf load_b(const __bf16* __restrict__ baseT, int ld,
                                        int ncol, int kb, int r, int h) {
  const __bf16* p = baseT + (size_t)(ncol + r) * (size_t)ld + kb + h * 16;
  return *(const v16bf*)p;
}

// --------------------------- prep kernels ----------------------------------

// src: (K,512) fp32 row-major  ->  dst: (512,K) bf16 row-major (one gate slice)
__global__ void transpose_w_bf16(const float* __restrict__ src,
                                 __bf16* __restrict__ dst, int K) {
  int idx = blockIdx.x * blockDim.x + threadIdx.x;
  if (idx >= 512 * K) return;
  int k = idx % K, n = idx / K;
  dst[idx] = (__bf16)src[(size_t)k * 512 + n];
}

// inputs (B,T,512) fp32 -> (T*B,512) bf16 time-major
__global__ void convert_x0_bf16(const float* __restrict__ in,
                                __bf16* __restrict__ out) {
  int idx = blockIdx.x * blockDim.x + threadIdx.x;
  if (idx >= NBATCH * T_SEQ * 512) return;
  int i = idx % 512;
  int b = (idx / 512) % NBATCH;
  int t = idx / (512 * NBATCH);
  out[idx] = (__bf16)in[((size_t)b * T_SEQ + t) * 512 + i];
}

// --------------------------- Phase A GEMM ----------------------------------
// G[dir] = X(bf16, M=16384 x K) @ WxT[dir]^T + bias[dir], fp32 out (M x 1536)
__global__ void __launch_bounds__(256)
gemm_xproj(const __bf16* __restrict__ X, int K,
           const __bf16* __restrict__ BT0, const __bf16* __restrict__ BT1,
           const float* __restrict__ bias0, const float* __restrict__ bias1,
           float* __restrict__ G0, float* __restrict__ G1) {
  const int lane = threadIdx.x & 31;
  const int wave = threadIdx.x >> 5;
  const int r = lane & 15, hh = lane >> 4;
  const bool d1 = (blockIdx.y != 0);
  const __bf16* BT = d1 ? BT1 : BT0;
  const float* bias = d1 ? bias1 : bias0;
  float* G = d1 ? G1 : G0;

  int tile = blockIdx.x * 8 + wave;        // 24576 tiles total
  int tm = (tile / 48) * 32;               // M = 16384 -> 512 row tiles
  int tn = (tile % 48) * 32;               // N = 1536  -> 48 col tiles

  v8f acc00 = {}, acc01 = {}, acc10 = {}, acc11 = {};
  for (int kb = 0; kb < K; kb += 32) {
    v16bf a0 = load_a(X, K, tm, kb, r, hh);
    v16bf a1 = load_a(X, K, tm + 16, kb, r, hh);
    v16bf b0 = load_b(BT, K, tn, kb, r, hh);
    v16bf b1 = load_b(BT, K, tn + 16, kb, r, hh);
    acc00 = wmma_bf16(a0, b0, acc00);
    acc01 = wmma_bf16(a0, b1, acc01);
    acc10 = wmma_bf16(a1, b0, acc10);
    acc11 = wmma_bf16(a1, b1, acc11);
  }
#pragma unroll
  for (int v = 0; v < 8; v++) {
    int m0 = tm + hh * 8 + v, m1 = m0 + 16;
    int n0 = tn + r, n1 = n0 + 16;
    G[(size_t)m0 * 1536 + n0] = acc00[v] + bias[n0];
    G[(size_t)m0 * 1536 + n1] = acc01[v] + bias[n1];
    G[(size_t)m1 * 1536 + n0] = acc10[v] + bias[n0];
    G[(size_t)m1 * 1536 + n1] = acc11[v] + bias[n1];
  }
}

// --------------------------- Phase B: persistent scan ----------------------
// blockIdx.x = direction; 1024 threads = 32 waves; whole T-loop in-kernel.
// Dynamic LDS: [0,64K) h fp32 | [64K,128K) z fp32 | [128K,160K) h bf16 |
//              [160K,192K) r*h bf16.
#define SCAN_LDS_BYTES (192 * 1024)

__global__ void __launch_bounds__(1024)
gru_scan(const float* __restrict__ G0, const float* __restrict__ G1,
         const __bf16* __restrict__ WhT0, const __bf16* __restrict__ WhT1,
         float* __restrict__ out, __bf16* __restrict__ out_bf,
         float* __restrict__ hfinal) {
  extern __shared__ char smem[];
  float*  h_s  = (float*)smem;                 // 64KB
  float*  z_s  = (float*)(smem + 65536);       // 64KB
  __bf16* hb_s = (__bf16*)(smem + 131072);     // 32KB
  __bf16* rh_s = (__bf16*)(smem + 163840);     // 32KB

  const int dir = blockIdx.x;
  const float* G = dir ? G1 : G0;
  const __bf16* WhT = dir ? WhT1 : WhT0;
  const int tid = threadIdx.x;
  const int lane = tid & 31;
  const int wave = tid >> 5;
  const int r = lane & 15, hh = lane >> 4;

  for (int i = tid; i < NBATCH * HID; i += 1024) {
    h_s[i] = 0.0f;
    hb_s[i] = (__bf16)0.0f;
  }
  __syncthreads();

  for (int t = 0; t < T_SEQ; t++) {
    const int tb = dir ? (T_SEQ - 1 - t) : t;
    const float* Gt = G + (size_t)tb * NBATCH * 1536;

    // ---- S1: r,z gates. N=1024, each wave 32 cols ----
    {
      int tn = wave * 32;
      v8f acc00 = {}, acc01 = {}, acc10 = {}, acc11 = {};
#pragma unroll 4
      for (int kb = 0; kb < HID; kb += 32) {
        v16bf a0 = load_a(hb_s, HID, 0, kb, r, hh);
        v16bf a1 = load_a(hb_s, HID, 16, kb, r, hh);
        v16bf b0 = load_b(WhT, HID, tn, kb, r, hh);
        v16bf b1 = load_b(WhT, HID, tn + 16, kb, r, hh);
        acc00 = wmma_bf16(a0, b0, acc00);
        acc01 = wmma_bf16(a0, b1, acc01);
        acc10 = wmma_bf16(a1, b0, acc10);
        acc11 = wmma_bf16(a1, b1, acc11);
      }
#pragma unroll
      for (int v = 0; v < 8; v++) {
#pragma unroll
        for (int q = 0; q < 4; q++) {
          int mi = q >> 1, ni = q & 1;
          int b_i = mi * 16 + hh * 8 + v;
          int n = tn + ni * 16 + r;
          float accv = (q == 0) ? acc00[v] : (q == 1) ? acc01[v]
                       : (q == 2) ? acc10[v] : acc11[v];
          float pre = accv + Gt[(size_t)b_i * 1536 + n];
          float s = 1.0f / (1.0f + __expf(-pre));
          if (n < HID)
            rh_s[b_i * HID + n] = (__bf16)(s * h_s[b_i * HID + n]);  // r*h
          else
            z_s[b_i * HID + (n - HID)] = s;                          // z
        }
      }
    }
    __syncthreads();
    // ---- S2: h_tilda + update. N=512, each wave 16 cols ----
    {
      int tn = wave * 16;
      v8f acc0 = {}, acc1 = {};
#pragma unroll 4
      for (int kb = 0; kb < HID; kb += 32) {
        v16bf a0 = load_a(rh_s, HID, 0, kb, r, hh);
        v16bf a1 = load_a(rh_s, HID, 16, kb, r, hh);
        v16bf b0 = load_b(WhT, HID, 1024 + tn, kb, r, hh);
        acc0 = wmma_bf16(a0, b0, acc0);
        acc1 = wmma_bf16(a1, b0, acc1);
      }
#pragma unroll
      for (int v = 0; v < 8; v++) {
#pragma unroll
        for (int mi = 0; mi < 2; mi++) {
          int b_i = mi * 16 + hh * 8 + v;
          int n = tn + r;
          float accv = mi ? acc1[v] : acc0[v];
          float pre = accv + Gt[(size_t)b_i * 1536 + 1024 + n];
          float ht = tanhf(pre);
          float z = z_s[b_i * HID + n];
          float hp = h_s[b_i * HID + n];
          float hn = z * ht + (1.0f - z) * hp;
          h_s[b_i * HID + n] = hn;
          hb_s[b_i * HID + n] = (__bf16)hn;
          if (out)  // layer 1 -> d_out, (B,T,2H)
            out[((size_t)b_i * T_SEQ + tb) * (2 * HID) + dir * HID + n] = hn;
          if (out_bf)  // layer 0 -> next layer's GEMM input, (T,B,2H) bf16
            out_bf[((size_t)tb * NBATCH + b_i) * (2 * HID) + dir * HID + n] =
                (__bf16)hn;
        }
      }
    }
    __syncthreads();
  }

  for (int i = tid; i < NBATCH * HID; i += 1024)
    hfinal[(size_t)dir * NBATCH * HID + i] = h_s[i];
}

// --------------------------- host orchestration ----------------------------
extern "C" void kernel_launch(void* const* d_in, const int* in_sizes, int n_in,
                              void* d_out, int out_size, void* d_ws,
                              size_t ws_size, hipStream_t stream) {
  (void)n_in; (void)out_size; (void)ws_size;
  const float* x_in = (const float*)d_in[0];

  // Param flattening order: insertion (Wxr,Whr,br,Wxz,Whz,bz,Wxh,Whh,bh) vs
  // jax-sorted (Whh,Whr,Whz,Wxh,Wxr,Wxz,bh,br,bz). Detect via size of entry 3.
  bool insertion = (in_sizes[3] == 512);
  int mapWx[3], mapWh[3], mapB[3];  // gate order r,z,h
  if (insertion) {
    mapWx[0] = 0; mapWx[1] = 3; mapWx[2] = 6;
    mapWh[0] = 1; mapWh[1] = 4; mapWh[2] = 7;
    mapB[0]  = 2; mapB[1]  = 5; mapB[2]  = 8;
  } else {
    mapWx[0] = 4; mapWx[1] = 5; mapWx[2] = 3;
    mapWh[0] = 1; mapWh[1] = 2; mapWh[2] = 0;
    mapB[0]  = 7; mapB[1]  = 8; mapB[2]  = 6;
  }
  auto cellp = [&](int c, int j) { return (const float*)d_in[1 + c * 9 + j]; };

  // workspace carve-up
  char* ws = (char*)d_ws;
  size_t off = 0;
  auto alloc = [&](size_t bytes) {
    size_t o = off;
    off += (bytes + 255) & ~(size_t)255;
    return o;
  };
  int DinC[4] = {512, 512, 1024, 1024};
  __bf16* WxT[4];
  __bf16* WhT[4];
  float* biasw[4];
  for (int c = 0; c < 4; c++)
    WxT[c] = (__bf16*)(ws + alloc((size_t)1536 * DinC[c] * 2));
  for (int c = 0; c < 4; c++)
    WhT[c] = (__bf16*)(ws + alloc((size_t)1536 * 512 * 2));
  for (int c = 0; c < 4; c++)
    biasw[c] = (float*)(ws + alloc(1536 * 4));
  __bf16* xin0 = (__bf16*)(ws + alloc((size_t)T_SEQ * NBATCH * 512 * 2));
  __bf16* xin1 = (__bf16*)(ws + alloc((size_t)T_SEQ * NBATCH * 1024 * 2));
  float* G0 = (float*)(ws + alloc((size_t)T_SEQ * NBATCH * 1536 * 4));
  float* G1 = (float*)(ws + alloc((size_t)T_SEQ * NBATCH * 1536 * 4));

  // allow 192KB dynamic LDS for the persistent scan kernel
  (void)hipFuncSetAttribute((const void*)gru_scan,
                            hipFuncAttributeMaxDynamicSharedMemorySize,
                            SCAN_LDS_BYTES);

  // weight convert + transpose to (N x K) bf16; bias concat via d2d copies
  for (int c = 0; c < 4; c++) {
    int K = DinC[c];
    for (int g = 0; g < 3; g++) {
      int totX = 512 * K;
      transpose_w_bf16<<<(totX + 255) / 256, 256, 0, stream>>>(
          cellp(c, mapWx[g]), WxT[c] + (size_t)g * 512 * K, K);
      transpose_w_bf16<<<(512 * 512 + 255) / 256, 256, 0, stream>>>(
          cellp(c, mapWh[g]), WhT[c] + (size_t)g * 512 * 512, 512);
      hipMemcpyAsync(biasw[c] + g * 512, cellp(c, mapB[g]), 512 * 4,
                     hipMemcpyDeviceToDevice, stream);
    }
  }
  {
    int tot = NBATCH * T_SEQ * 512;
    convert_x0_bf16<<<(tot + 255) / 256, 256, 0, stream>>>(x_in, xin0);
  }

  const size_t OUT_ELEMS = (size_t)NBATCH * T_SEQ * 2 * HID;
  for (int layer = 0; layer < 2; layer++) {
    int c0 = layer * 2, c1 = c0 + 1;
    const __bf16* X = (layer == 0) ? xin0 : xin1;
    int K = (layer == 0) ? 512 : 1024;

    gemm_xproj<<<dim3(512 * 48 / 8, 2), 256, 0, stream>>>(
        X, K, WxT[c0], WxT[c1], biasw[c0], biasw[c1], G0, G1);

    float*  outp  = (layer == 1) ? (float*)d_out : nullptr;
    __bf16* outbf = (layer == 0) ? xin1 : nullptr;
    gru_scan<<<dim3(2), 1024, SCAN_LDS_BYTES, stream>>>(
        G0, G1, WhT[c0], WhT[c1], outp, outbf,
        (float*)d_out + OUT_ELEMS + (size_t)layer * 2 * NBATCH * HID);
  }
}
